// NextSequencePredictor_v4_32461362823774
// MI455X (gfx1250) — matrix operand order repaired and checked by reading, not compiled, not run on hardware
//
#include <hip/hip_runtime.h>
#include <hip/hip_bf16.h>
#include <cstdint>

typedef __attribute__((ext_vector_type(16))) _Float16 v16h;
typedef __attribute__((ext_vector_type(8)))  _Float16 v8h;
typedef __attribute__((ext_vector_type(8)))  float    v8f;

// ---------------------------------------------------------------------------
// Implicit-GEMM conv (normal SAME conv / stride-2 transposed conv) on the
// CDNA5 WMMA pipe.  K-space: k = tap*Cpad + ci, Cpad = max(32,Ci) (power of
// two for every layer) so each 32-wide K block lies in ONE filter tap:
// tap/ky/kx + boundary test are computed once per K step and per-lane A data
// is two contiguous 8-element runs (vector loads).  Weights are pre-packed
// to fragment-ordered f16: dst[((nt*KB+kb)*32+lane)*16+h] with
//   k = kb*32 + (lane>=16 ? 8 : 0) + (h&7) + (h>=8 ? 16 : 0),  n = nt*16 + (lane&15)
// matching the CDNA5 B (32x16 f16) VGPR layout.  A fragment (16x32 f16):
//   lane<16 : M=lane,    h0..7 = K+0..7,  h8..15 = K+16..23
//   lane>=16: M=lane-16, h0..7 = K+8..15, h8..15 = K+24..31
// C/D (16x16 f32): element r -> row (lane>>4)*8 + r, col lane&15.
// Per-wave tile: 32(M) x 64(N) = 8 WMMAs per K step.
// Activations are stored f16 (except f32 network input / gate pre-activations
// / final output) to halve HBM traffic and drop inner-loop cvt ops.
// ---------------------------------------------------------------------------

struct ConvP {
  int NF;            // frames
  int Hi, Wi, Ci;    // input dims
  int Ho, Wo, Co;    // output dims
  int stride;        // normal conv stride
  int transpose;     // 1 = stride-2 transposed conv (gather form)
  int padBeg;        // SAME pad-begin (0 for s=2 even input, 1 for s=1)
  int act;           // 1 = ReLU before BN
  int shift;         // log2(Cpad)
  int KB;            // (9*Cpad)/32
  long inFS, outFS;  // per-frame element strides
};

template <typename TI, typename TO>
__global__ __launch_bounds__(32)
void conv_wmma(const TI* __restrict__ X, const _Float16* __restrict__ Wp,
               const float* __restrict__ bias,
               const float* __restrict__ gptr, const float* __restrict__ betap,
               const float* __restrict__ meanp, const float* __restrict__ varp,
               TO* __restrict__ Y, ConvP p)
{
  const int lane  = threadIdx.x;
  const int HW    = p.Ho * p.Wo;
  const int M     = p.NF * HW;
  const int cmask = (1 << p.shift) - 1;

  // Two 16-row M sub-tiles handled by this wave
  const int maBase = (int)blockIdx.x * 32;
  long pb[2]; int oy[2], ox[2]; bool mv[2];
#pragma unroll
  for (int mt = 0; mt < 2; ++mt) {
    const int ma = maBase + mt * 16 + (lane & 15);
    mv[mt] = ma < M;
    int an = 0, pix = 0;
    if (mv[mt]) { an = ma / HW; pix = ma - an * HW; }
    oy[mt] = pix / p.Wo;
    ox[mt] = pix - oy[mt] * p.Wo;
    pb[mt] = (long)an * p.inFS;
  }

  // Four 16-col N sub-tiles
  const int co0 = (int)blockIdx.y * 64 + (lane & 15);
  v8f acc[2][4];
#pragma unroll
  for (int j = 0; j < 4; ++j) {
    const int co = co0 + j * 16;
    const float b0 = (bias && co < p.Co) ? bias[co] : 0.0f;
#pragma unroll
    for (int r = 0; r < 8; ++r) { acc[0][j][r] = b0; acc[1][j][r] = b0; }
  }

  const int khA = (lane >> 4) << 3;
  const _Float16* wbase =
      Wp + (((long)((int)blockIdx.y * 4) * p.KB) * 32 + lane) * 16;

  for (int kb = 0; kb < p.KB; ++kb) {
    const int k0  = kb * 32;
    const int tap = k0 >> p.shift;          // scalar: whole block = one tap
    const int ky  = tap / 3;
    const int kx  = tap - 3 * ky;
    const int cib = (k0 & cmask) + khA;

    if (kb + 1 < p.KB)
      __builtin_prefetch(wbase + (long)(kb + 1) * 512, 0, 3);

    v16h Af[2];
#pragma unroll
    for (int mt = 0; mt < 2; ++mt) {
      int iy, ix;
      bool ok;
      if (p.transpose) {
        const int sy = oy[mt] + ky, sx = ox[mt] + kx;
        ok = ((sy | sx) & 1) == 0;          // both even
        iy = (sy >> 1) - 1;
        ix = (sx >> 1) - 1;
        ok = ok && iy >= 0 && iy < p.Hi && ix >= 0 && ix < p.Wi;
      } else {
        iy = oy[mt] * p.stride + ky - p.padBeg;
        ix = ox[mt] * p.stride + kx - p.padBeg;
        ok = iy >= 0 && iy < p.Hi && ix >= 0 && ix < p.Wi;
      }
      ok = ok && mv[mt];

      v16h a;
      if (ok) {
        const TI* ap = X + pb[mt] + ((long)iy * p.Wi + ix) * p.Ci + cib;
        if (p.Ci >= 32) {
          if constexpr (sizeof(TI) == 2) {
            // f16 activations: two 16-byte loads, no conversion
            const v8h f0 = *(const v8h*)(ap);
            const v8h f1 = *(const v8h*)(ap + 16);
#pragma unroll
            for (int j2 = 0; j2 < 8; ++j2) { a[j2] = f0[j2]; a[8 + j2] = f1[j2]; }
          } else {
            const float4 f0 = *(const float4*)(ap);
            const float4 f1 = *(const float4*)(ap + 4);
            const float4 f2 = *(const float4*)(ap + 16);
            const float4 f3 = *(const float4*)(ap + 20);
            a[0]  = (_Float16)f0.x; a[1]  = (_Float16)f0.y;
            a[2]  = (_Float16)f0.z; a[3]  = (_Float16)f0.w;
            a[4]  = (_Float16)f1.x; a[5]  = (_Float16)f1.y;
            a[6]  = (_Float16)f1.z; a[7]  = (_Float16)f1.w;
            a[8]  = (_Float16)f2.x; a[9]  = (_Float16)f2.y;
            a[10] = (_Float16)f2.z; a[11] = (_Float16)f2.w;
            a[12] = (_Float16)f3.x; a[13] = (_Float16)f3.y;
            a[14] = (_Float16)f3.z; a[15] = (_Float16)f3.w;
          }
        } else {
          // Ci < 32 (padded channels): guarded scalar loads
#pragma unroll
          for (int j2 = 0; j2 < 8; ++j2) {
            const int c1 = cib + j2, c2 = cib + 16 + j2;
            a[j2]     = (c1 < p.Ci) ? (_Float16)ap[j2]      : (_Float16)0.0f;
            a[8 + j2] = (c2 < p.Ci) ? (_Float16)ap[16 + j2] : (_Float16)0.0f;
          }
        }
      } else {
#pragma unroll
        for (int j2 = 0; j2 < 16; ++j2) a[j2] = (_Float16)0.0f;
      }
      Af[mt] = a;
    }

#pragma unroll
    for (int j = 0; j < 4; ++j) {
      const v16h Bf = *(const v16h*)(wbase + ((long)j * p.KB + kb) * 512);
      acc[0][j] = __builtin_amdgcn_wmma_f32_16x16x32_f16(
          false, Af[0], false, Bf, (short)0, acc[0][j], false, false);
      acc[1][j] = __builtin_amdgcn_wmma_f32_16x16x32_f16(
          false, Af[1], false, Bf, (short)0, acc[1][j], false, false);
    }
  }

  // Epilogue: ReLU + inference BN, scatter NHWC with per-frame stride
  const int mb = (lane >> 4) << 3;
#pragma unroll
  for (int mt = 0; mt < 2; ++mt) {
#pragma unroll
    for (int j = 0; j < 4; ++j) {
      const int co = co0 + j * 16;
      if (co < p.Co) {
        float gg = 1.0f, bb = 0.0f, mm = 0.0f, vv = 1.0f;
        if (gptr) { gg = gptr[co]; bb = betap[co]; mm = meanp[co]; vv = varp[co]; }
        const float rs = rsqrtf(vv + 1e-3f);
#pragma unroll
        for (int r = 0; r < 8; ++r) {
          const int mg = maBase + mt * 16 + mb + r;
          if (mg < M) {
            float v = acc[mt][j][r];
            if (p.act) v = fmaxf(v, 0.0f);
            if (gptr) v = gg * (v - mm) * rs + bb;
            const int n2  = mg / HW;
            const int pix = mg - n2 * HW;
            Y[(long)n2 * p.outFS + (long)pix * p.Co + co] = (TO)v;
          }
        }
      }
    }
  }
}

// ---------------------------------------------------------------------------
// Dense: Y = BN(relu(X @ W + b)); X:[M,K] f32, Wp packed f16 (K padded to 32)
// ---------------------------------------------------------------------------
template <typename TO>
__global__ __launch_bounds__(32)
void dense_wmma(const float* __restrict__ X, const _Float16* __restrict__ Wp,
                const float* __restrict__ bias,
                const float* __restrict__ gptr, const float* __restrict__ betap,
                const float* __restrict__ meanp, const float* __restrict__ varp,
                TO* __restrict__ Y, int M, int K, int KB, int N)
{
  const int lane = threadIdx.x;
  const int  ma  = (int)blockIdx.x * 16 + (lane & 15);
  const bool mav = ma < M;
  const int  co0 = (int)blockIdx.y * 64 + (lane & 15);

  v8f acc[4];
#pragma unroll
  for (int j = 0; j < 4; ++j) {
    const int co = co0 + j * 16;
    const float b0 = (co < N) ? bias[co] : 0.0f;
#pragma unroll
    for (int r = 0; r < 8; ++r) acc[j][r] = b0;
  }

  const int khA = (lane >> 4) << 3;

  for (int kb = 0; kb < KB; ++kb) {
    const int kbase = kb * 32 + khA;
    v16h Af;
#pragma unroll
    for (int j2 = 0; j2 < 8; ++j2) {
      const int k1 = kbase + j2, k2 = kbase + 16 + j2;
      Af[j2]     = (_Float16)((mav && k1 < K) ? X[(long)ma * K + k1] : 0.0f);
      Af[8 + j2] = (_Float16)((mav && k2 < K) ? X[(long)ma * K + k2] : 0.0f);
    }
#pragma unroll
    for (int j = 0; j < 4; ++j) {
      const v16h Bf = *(const v16h*)(
          Wp + ((((long)((int)blockIdx.y * 4 + j) * KB + kb) << 5) + lane) * 16);
      acc[j] = __builtin_amdgcn_wmma_f32_16x16x32_f16(
          false, Af, false, Bf, (short)0, acc[j], false, false);
    }
  }

  const int mb = (lane >> 4) << 3;
#pragma unroll
  for (int j = 0; j < 4; ++j) {
    const int co = co0 + j * 16;
    if (co < N) {
      const float gg = gptr[co], bb = betap[co], mm = meanp[co], vv = varp[co];
      const float rs = rsqrtf(vv + 1e-3f);
#pragma unroll
      for (int r = 0; r < 8; ++r) {
        const int mg = (int)blockIdx.x * 16 + mb + r;
        if (mg < M) {
          float v = fmaxf(acc[j][r], 0.0f);
          Y[(long)mg * N + co] = (TO)(gg * (v - mm) * rs + bb);
        }
      }
    }
  }
}

// ---------------------------------------------------------------------------
// Weight packer: f32 HWIO (or dense [K,N]) -> fragment-ordered f16.
// mode 0 (conv): tap = k>>shift, ci = k&cmask, src (tap*Ci+ci)*Co+n, ci<Ci
// mode 1 (dense): src k*Co+n, k<Ci (Ci carries real K)
// ---------------------------------------------------------------------------
__global__ void pack_w(const float* __restrict__ Wsrc, _Float16* __restrict__ dst,
                       int Ci, int cmask, int shift, int Co, int KB, int mode,
                       long total)
{
  const long idx = (long)blockIdx.x * blockDim.x + threadIdx.x;
  if (idx >= total) return;
  const int h    = (int)(idx & 15);
  const int lane = (int)((idx >> 4) & 31);
  const long rest = idx >> 9;
  const int kb = (int)(rest % KB);
  const int nt = (int)(rest / KB);
  const int k  = kb * 32 + ((lane >> 4) << 3) + (h & 7) + ((h >> 3) << 4);
  const int n  = nt * 16 + (lane & 15);
  float v = 0.0f;
  if (n < Co) {
    if (mode == 0) {
      const int ci  = k & cmask;
      const int tap = k >> shift;
      if (ci < Ci) v = Wsrc[((long)tap * Ci + ci) * Co + n];
    } else {
      if (k < Ci) v = Wsrc[(long)k * Co + n];
    }
  }
  dst[idx] = (_Float16)v;
}

// ---------------------------------------------------------------------------
__global__ void lstm_gate(const float* __restrict__ zx, const float* __restrict__ zh,
                          float* __restrict__ cst, _Float16* __restrict__ hst,
                          _Float16* __restrict__ yseq,
                          const float* __restrict__ g, const float* __restrict__ be,
                          const float* __restrict__ mu, const float* __restrict__ va,
                          int T, int t, int HW, int C, int total)
{
  const int i = blockIdx.x * blockDim.x + threadIdx.x;
  if (i >= total) return;
  const int c   = i % C;
  const int r   = i / C;
  const int pix = r % HW;
  const int b   = r / HW;
  const long base4 = ((long)b * HW + pix) * 4 * C + c;

  const float zi = zx[base4]         + zh[base4];
  const float zf = zx[base4 + C]     + zh[base4 + C];
  const float zg = zx[base4 + 2 * C] + zh[base4 + 2 * C];
  const float zo = zx[base4 + 3 * C] + zh[base4 + 3 * C];

  const float hi = fminf(fmaxf(0.2f * zi + 0.5f, 0.0f), 1.0f);
  const float hf = fminf(fmaxf(0.2f * zf + 0.5f, 0.0f), 1.0f);
  const float ho = fminf(fmaxf(0.2f * zo + 0.5f, 0.0f), 1.0f);

  const float cn = hf * cst[i] + hi * tanhf(zg);
  const float hn = ho * tanhf(cn);
  cst[i] = cn;
  hst[i] = (_Float16)hn;

  const float y = g[c] * (hn - mu[c]) * rsqrtf(va[c] + 1e-3f) + be[c];
  yseq[((long)(b * T + t) * HW + pix) * C + c] = (_Float16)y;
}

__global__ void fill0(float* __restrict__ p, int n)
{
  const int i = blockIdx.x * blockDim.x + threadIdx.x;
  if (i < n) p[i] = 0.0f;
}

__global__ void fillh(_Float16* __restrict__ p, int n)
{
  const int i = blockIdx.x * blockDim.x + threadIdx.x;
  if (i < n) p[i] = (_Float16)0.0f;
}

__global__ void concat_fan(const _Float16* __restrict__ y7,
                           const float* __restrict__ fan,
                           float* __restrict__ out, int perb, int flat, int total)
{
  const int i = blockIdx.x * blockDim.x + threadIdx.x;
  if (i >= total) return;
  const int b   = i / flat;
  const int col = i - b * flat;
  out[i] = (col < perb) ? (float)y7[(long)b * perb + col]
                        : fan[b * 10 + (col - perb)];
}

// ---------------------------------------------------------------------------

template <typename TI, typename TO>
static inline void launch_conv(const TI* X, const _Float16* Wp, const float* bias,
                               const float* g, const float* be, const float* mu,
                               const float* va, TO* Y, const ConvP& p,
                               hipStream_t s)
{
  const int M = p.NF * p.Ho * p.Wo;
  dim3 grid((M + 31) / 32, (p.Co + 63) / 64);
  conv_wmma<TI, TO><<<grid, 32, 0, s>>>(X, Wp, bias, g, be, mu, va, Y, p);
}

extern "C" void kernel_launch(void* const* d_in, const int* in_sizes, int n_in,
                              void* d_out, int out_size, void* d_ws, size_t ws_size,
                              hipStream_t stream)
{
  (void)in_sizes; (void)n_in; (void)out_size; (void)ws_size;

  const int B = 2, T = 4;
  const float* Xseq = (const float*)d_in[0];
  const float* fan  = (const float*)d_in[1];

  // ---- workspace carving ----
  float* w = (float*)d_ws;
  const long SZ = 10485760;                  // max activation elem count
  _Float16* bufA = (_Float16*)w;             // f16 sequence / decoder buffers
  _Float16* bufB = (_Float16*)(w + SZ);
  float* zx   = w + 2 * SZ;                  // gate pre-activations (f32)
  float* zh   = w + 3 * SZ;
  _Float16* hst = (_Float16*)(w + 4 * SZ);   // hidden state (f16)
  float* cst  = w + 4 * SZ + 1310720;        // cell state (f32)
  float* cat  = cst + 2621440;               // [2, 40970] f32
  float* dd1  = cat + 81952;
  float* dd2  = dd1 + 256;
  _Float16* dd4 = (_Float16*)(dd2 + 1024);   // [2, 10240] f16 (decoder input)
  _Float16* wpk = (_Float16*)(dd2 + 1024 + 20480);  // packed weights (f16)
  long wcur = 0;

  auto packConv = [&](const float* Wsrc, int Ci, int Co, int& KBout,
                      int& shiftOut) -> const _Float16* {
    const int Cpad  = (Ci < 32) ? 32 : Ci;   // all Ci are powers of two
    const int shift = __builtin_ctz(Cpad);
    const int KB    = (9 * Cpad) / 32;
    const int ntall = ((Co + 63) / 64) * 4;
    const long total = (long)ntall * KB * 512;
    _Float16* dst = wpk + wcur;
    wcur += total;
    pack_w<<<(unsigned)((total + 255) / 256), 256, 0, stream>>>(
        Wsrc, dst, Ci, Cpad - 1, shift, Co, KB, 0, total);
    KBout = KB; shiftOut = shift;
    return dst;
  };
  auto packDense = [&](const float* Wsrc, int K, int N, int& KBout)
      -> const _Float16* {
    const int KB    = (K + 31) / 32;
    const int ntall = ((N + 63) / 64) * 4;
    const long total = (long)ntall * KB * 512;
    _Float16* dst = wpk + wcur;
    wcur += total;
    pack_w<<<(unsigned)((total + 255) / 256), 256, 0, stream>>>(
        Wsrc, dst, K, 0, 0, N, KB, 1, total);
    KBout = KB;
    return dst;
  };

  // ================= encoder: 7 ConvLSTM + BN =================
  int pi = 2;
  int Hi = 512, Wi = 640, Cin = 1;
  const int couts[7] = {16, 32, 64, 128, 256, 512, 512};

  for (int L = 0; L < 7; ++L) {
    const float* Wx = (const float*)d_in[pi++];
    const float* Wh = (const float*)d_in[pi++];
    const float* bv = (const float*)d_in[pi++];
    const float* bg = (const float*)d_in[pi++];
    const float* bb = (const float*)d_in[pi++];
    const float* bm = (const float*)d_in[pi++];
    const float* bV = (const float*)d_in[pi++];

    const int C  = couts[L];
    const int Ho = Hi / 2, Wo = Wi / 2;
    const int HW = Ho * Wo;

    int KBx, shx, KBh, shh;
    const _Float16* Wxp = packConv(Wx, Cin, 4 * C, KBx, shx);
    const _Float16* Whp = packConv(Wh, C, 4 * C, KBh, shh);

    const _Float16* src = (L & 1) ? bufA : bufB;  // L>0 input (f16)
    _Float16*       dst = (L & 1) ? bufB : bufA;

    const int nstate = B * HW * C;
    fillh<<<(nstate + 255) / 256, 256, 0, stream>>>(hst, nstate);
    fill0<<<(nstate + 255) / 256, 256, 0, stream>>>(cst, nstate);

    ConvP px;   // input conv, stride 2, SAME pad (0,1)
    px.NF = B; px.Hi = Hi; px.Wi = Wi; px.Ci = Cin;
    px.Ho = Ho; px.Wo = Wo; px.Co = 4 * C;
    px.stride = 2; px.transpose = 0; px.padBeg = 0; px.act = 0;
    px.shift = shx; px.KB = KBx;
    px.inFS = (long)T * Hi * Wi * Cin; px.outFS = (long)HW * 4 * C;

    ConvP ph;   // recurrent conv, stride 1, SAME pad (1,1)
    ph.NF = B; ph.Hi = Ho; ph.Wi = Wo; ph.Ci = C;
    ph.Ho = Ho; ph.Wo = Wo; ph.Co = 4 * C;
    ph.stride = 1; ph.transpose = 0; ph.padBeg = 1; ph.act = 0;
    ph.shift = shh; ph.KB = KBh;
    ph.inFS = (long)HW * C; ph.outFS = (long)HW * 4 * C;

    for (int t = 0; t < T; ++t) {
      if (L == 0)
        launch_conv<float, float>(Xseq + (long)t * Hi * Wi * Cin, Wxp, bv,
                                  nullptr, nullptr, nullptr, nullptr, zx, px,
                                  stream);
      else
        launch_conv<_Float16, float>(src + (long)t * Hi * Wi * Cin, Wxp, bv,
                                     nullptr, nullptr, nullptr, nullptr, zx, px,
                                     stream);
      launch_conv<_Float16, float>(hst, Whp, nullptr,
                                   nullptr, nullptr, nullptr, nullptr, zh, ph,
                                   stream);
      lstm_gate<<<(nstate + 255) / 256, 256, 0, stream>>>(
          zx, zh, cst, hst, dst, bg, bb, bm, bV, T, t, HW, C, nstate);
    }
    Hi = Ho; Wi = Wo; Cin = C;
  }

  // ================= dense bottleneck =================
  const _Float16* enc_out = bufA;          // layer index 6 (even) wrote bufA
  const int perb = T * 4 * 5 * 512;        // 40960
  const int flat = perb + 10;              // 40970
  {
    const int total = B * flat;
    concat_fan<<<(total + 255) / 256, 256, 0, stream>>>(
        enc_out, fan, cat, perb, flat, total);
  }

  for (int l = 0; l < 3; ++l) {
    static const int dK[3] = {0, 100, 512};     // d1 K patched below
    const int K = (l == 0) ? flat : dK[l];
    const int N = (l == 0) ? 100 : (l == 1 ? 512 : 10240);
    const float* Xd = (l == 0) ? cat : (l == 1 ? dd1 : dd2);

    const float* Wd = (const float*)d_in[pi++];
    const float* bd = (const float*)d_in[pi++];
    const float* bg = (const float*)d_in[pi++];
    const float* bb = (const float*)d_in[pi++];
    const float* bm = (const float*)d_in[pi++];
    const float* bV = (const float*)d_in[pi++];
    int KB;
    const _Float16* Wdp = packDense(Wd, K, N, KB);
    dim3 grid((B + 15) / 16, (N + 63) / 64);
    if (l == 2)
      dense_wmma<_Float16><<<grid, 32, 0, stream>>>(Xd, Wdp, bd, bg, bb, bm, bV,
                                                    dd4, B, K, KB, N);
    else
      dense_wmma<float><<<grid, 32, 0, stream>>>(Xd, Wdp, bd, bg, bb, bm, bV,
                                                 (l == 0) ? dd1 : dd2, B, K, KB,
                                                 N);
  }

  // ================= decoder: 6 Conv2DTranspose(relu)+BN =================
  int dh = 8, dw = 10, dci = 64;
  const int dcouts[6] = {64, 32, 16, 8, 4, 1};
  const _Float16* dsrc = dd4;              // [B*2, 8, 10, 64] f16 frames
  for (int L = 0; L < 6; ++L) {
    const float* Kd = (const float*)d_in[pi++];
    const float* bd = (const float*)d_in[pi++];
    const float* bg = (const float*)d_in[pi++];
    const float* bb = (const float*)d_in[pi++];
    const float* bm = (const float*)d_in[pi++];
    const float* bV = (const float*)d_in[pi++];

    const int co = dcouts[L];
    const int ho = 2 * dh, wo = 2 * dw;

    int KB, sh;
    const _Float16* Kp = packConv(Kd, dci, co, KB, sh);

    ConvP pt;
    pt.NF = B * 2; pt.Hi = dh; pt.Wi = dw; pt.Ci = dci;
    pt.Ho = ho; pt.Wo = wo; pt.Co = co;
    pt.stride = 2; pt.transpose = 1; pt.padBeg = 0; pt.act = 1;
    pt.shift = sh; pt.KB = KB;
    pt.inFS = (long)dh * dw * dci; pt.outFS = (long)ho * wo * co;

    if (L == 5) {
      launch_conv<_Float16, float>(dsrc, Kp, bd, bg, bb, bm, bV,
                                   (float*)d_out, pt, stream);
    } else {
      _Float16* ddst = (L & 1) ? bufB : bufA;
      launch_conv<_Float16, _Float16>(dsrc, Kp, bd, bg, bb, bm, bV, ddst, pt,
                                      stream);
      dsrc = ddst;
    }
    dh = ho; dw = wo; dci = co;
  }
}
